// GCNZ_2886218022957
// MI455X (gfx1250) — matrix operand order repaired and checked by reading, not compiled, not run on hardware
//
#include <hip/hip_runtime.h>

// ---------------------------------------------------------------------------
// GCN 3-layer pipeline for MI455X (gfx1250, wave32, WMMA).
//  - GEMMs in bf16 via v_wmma_f32_16x16x32_bf16 (f32 accumulate), K-step 64
//  - X tile staged with GLOBAL_LOAD_ASYNC_TO_LDS_B128 (ASYNCcnt path)
//  - W tile transposed into LDS with packed b32 stores (K-pairs contiguous)
//  - Edge aggregation with f32 global atomics (H fits in 192MB L2)
//  - BatchNorm (training-mode, biased var) + ReLU fused, emits bf16
// ---------------------------------------------------------------------------

typedef __bf16 bf16_t;
typedef __attribute__((ext_vector_type(16))) __bf16 v16bf;
typedef __attribute__((ext_vector_type(8)))  float  v8f;

#define BN_EPS 1e-5f

static __device__ __forceinline__ void atomAddF(float* p, float v) {
  __hip_atomic_fetch_add(p, v, __ATOMIC_RELAXED, __HIP_MEMORY_SCOPE_AGENT);
}

// ------------------------------- utility ----------------------------------

__global__ void k_fill_f32(float* __restrict__ p, float v, long long n) {
  long long i  = (long long)blockIdx.x * blockDim.x + threadIdx.x;
  long long st = (long long)gridDim.x * blockDim.x;
  for (; i < n; i += st) p[i] = v;
}

__global__ void k_deg(const int* __restrict__ dstv, float* __restrict__ deg, int E) {
  int i = blockIdx.x * blockDim.x + threadIdx.x;
  if (i < E) atomAddF(&deg[dstv[i]], 1.0f);
}

__global__ void k_rsqrt_ip(float* __restrict__ p, int n) {
  int i = blockIdx.x * blockDim.x + threadIdx.x;
  if (i < n) p[i] = rsqrtf(p[i]);   // deg >= 1 always (self-loop)
}

__global__ void k_cvt_bf16(const float* __restrict__ x, bf16_t* __restrict__ y, long long n) {
  long long i  = (long long)blockIdx.x * blockDim.x + threadIdx.x;
  long long st = (long long)gridDim.x * blockDim.x;
  for (; i < n; i += st) y[i] = (bf16_t)x[i];
}

// Convert K x Nc f32 -> K x NcP bf16, zero-padding columns Nc..NcP-1.
__global__ void k_cvt_bf16_pad(const float* __restrict__ x, bf16_t* __restrict__ y,
                               int Nc, int NcP, long long totalP) {
  long long i  = (long long)blockIdx.x * blockDim.x + threadIdx.x;
  long long st = (long long)gridDim.x * blockDim.x;
  for (; i < totalP; i += st) {
    int k = (int)(i / NcP), n = (int)(i % NcP);
    y[i] = (n < Nc) ? (bf16_t)x[(long long)k * Nc + n] : (bf16_t)0.0f;
  }
}

__global__ void k_bias_init(const float* __restrict__ b, float* __restrict__ out,
                            int F, long long total) {
  long long i  = (long long)blockIdx.x * blockDim.x + threadIdx.x;
  long long st = (long long)gridDim.x * blockDim.x;
  for (; i < total; i += st) out[i] = b[(int)(i % F)];
}

// ------------------------------ WMMA GEMM ---------------------------------
// Y[M x Ncol] = X[M x K] (bf16) @ W[K x Ncol] (bf16), f32 accumulate.
// Requirements (enforced by driver): M % 32 == 0, K % 64 == 0, Ncol % 64 == 0.
// Block: 256 threads = 8 waves; block tile 32 rows x 64 cols; wave -> 16x16.
// K-step 64: two v_wmma per barrier pair.

__global__ void k_gemm_bf16_full(const bf16_t* __restrict__ X, const bf16_t* __restrict__ W,
                                 float* __restrict__ Y, int K, int Ncol) {
  __shared__ __align__(16) bf16_t sX[32][64];   // [row][k]
  __shared__ __align__(16) bf16_t sW[64][64];   // [col][k] transposed: K-pairs contiguous

  const int tid  = threadIdx.x;
  const int wave = tid >> 5;
  const int lane = tid & 31;
  const int m0 = blockIdx.y * 32;
  const int n0 = blockIdx.x * 64;
  const int mr = (wave & 1) * 16;     // wave row offset in tile
  const int nc = (wave >> 1) * 16;    // wave col offset in tile
  const int lm = lane & 15;
  const int kg = lane >> 4;           // half-wave K group

  // staging coordinates
  const int xr  = (tid * 8) >> 6;     // 0..31  (X row)
  const int xk  = (tid * 8) & 63;     // 0,8,..,56 (X k offset)
  const int wk2 = (tid >> 3) * 2;     // 0,2,..,62 (W k row pair)
  const int wn  = (tid & 7) * 8;      // 0,8,..,56 (W col offset)

  v8f acc = {0.f, 0.f, 0.f, 0.f, 0.f, 0.f, 0.f, 0.f};

  const bf16_t* xrow = X + (long long)(m0 + xr) * K;
  // 32-bit LDS offset of this thread's X destination (shared-aperture low bits).
  const unsigned lds_x = (unsigned)(unsigned long long)&sX[xr][xk];
  unsigned* const sWp = (unsigned*)&sW[0][0];

  for (int k0 = 0; k0 < K; k0 += 64) {
    // ---- stage X: async global->LDS, 16B per lane (ASYNCcnt) ----
    {
      unsigned long long ga = (unsigned long long)(xrow + k0 + xk);
      asm volatile("global_load_async_to_lds_b128 %0, %1, off"
                   :: "v"(lds_x), "v"(ga) : "memory");
    }
    // ---- stage W transposed: two adjacent K rows, pack (k,k+1) -> b32 ----
    {
      const bf16_t* wp = W + (long long)(k0 + wk2) * Ncol + n0 + wn;
      uint4 lo = *(const uint4*)wp;
      uint4 hi = *(const uint4*)(wp + Ncol);
      const unsigned short* plo = (const unsigned short*)&lo;
      const unsigned short* phi = (const unsigned short*)&hi;
      const int base = wn * 32 + (wk2 >> 1);   // uint-granular index into sW
      for (int i = 0; i < 8; ++i) {
        sWp[base + i * 32] = (unsigned)plo[i] | ((unsigned)phi[i] << 16);
      }
    }
    // prefetch next K-tile while we compute (global_prefetch_b8)
    if (k0 + 64 < K) {
      __builtin_prefetch(xrow + k0 + 64 + xk, 0, 1);
      __builtin_prefetch(W + (long long)(k0 + 64 + wk2) * Ncol + n0 + wn, 0, 1);
    }
    // async X copy must land in LDS before the barrier releases readers
    asm volatile("s_wait_asynccnt 0x0" ::: "memory");
    __syncthreads();

    // ---- two WMMAs over the 64-deep K tile ----
    for (int kh = 0; kh < 2; ++kh) {
      const int kb = kh * 32;
      v16bf a, b;
      // A (16x32): lane m = lm; VGPR j(0..3): K = kg*8 + 2j; VGPR 4..7: +16.
      for (int j = 0; j < 4; ++j) {
        int k = kb + kg * 8 + 2 * j;
        a[2 * j]         = sX[mr + lm][k];
        a[2 * j + 1]     = sX[mr + lm][k + 1];
        a[8 + 2 * j]     = sX[mr + lm][k + 16];
        a[8 + 2 * j + 1] = sX[mr + lm][k + 17];
      }
      // B (32x16): lane n = lm; VGPR j(0..3): K = kg*8 + 2j; VGPR 4..7: +16.
      for (int j = 0; j < 8; ++j) {
        int k = kb + kg * 8 + (j & 3) * 2 + (j >> 2) * 16;
        b[2 * j]     = sW[nc + lm][k];
        b[2 * j + 1] = sW[nc + lm][k + 1];
      }
      acc = __builtin_amdgcn_wmma_f32_16x16x32_bf16(
          /*neg_a=*/false, a, /*neg_b=*/false, b,
          /*c_mod=*/(short)0, acc, /*reuse_a=*/false, /*reuse_b=*/false);
    }
    __syncthreads();
  }

  // Store D: VGPR r holds M = r + kg*8, N = lm (C/D layout table).
  const int col = n0 + nc + lm;
  float* yrow = Y + (long long)(m0 + mr + kg * 8) * Ncol + col;
  for (int r = 0; r < 8; ++r) {
    yrow[(long long)r * Ncol] = acc[r];
  }
}

// ---------------------------- aggregation ---------------------------------
// out[dst*F + c] += h[src*ldh + c] * dinv[src]*dinv[dst] over E edges + N
// self-loops; c in [0, F), F % 4 == 0. ldh may exceed F (padded layer 3).

__global__ void k_aggregate(const float* __restrict__ h, const int* __restrict__ src,
                            const int* __restrict__ dstv, const float* __restrict__ dinv,
                            float* __restrict__ out, int F, int ldh, int E, int Nn) {
  const int C = F >> 2;
  const long long total = (long long)(E + Nn) * C;
  long long i  = (long long)blockIdx.x * blockDim.x + threadIdx.x;
  const long long st = (long long)gridDim.x * blockDim.x;
  for (; i < total; i += st) {
    int e  = (int)(i / C);
    int c4 = ((int)(i % C)) << 2;
    int s, d;
    if (e < E) { s = src[e]; d = dstv[e]; } else { s = e - E; d = s; }
    float nrm = dinv[s] * dinv[d];
    float4 hv = *(const float4*)(h + (long long)s * ldh + c4);
    float* op = out + (long long)d * F + c4;
    atomAddF(op + 0, hv.x * nrm);
    atomAddF(op + 1, hv.y * nrm);
    atomAddF(op + 2, hv.z * nrm);
    atomAddF(op + 3, hv.w * nrm);
  }
}

// ------------------------------ batchnorm ---------------------------------

__global__ void k_bn_stats(const float* __restrict__ x, float* __restrict__ sums,
                           float* __restrict__ sumsq, int Nrows, int F, int rowsPerBlk) {
  int c = blockIdx.x * blockDim.x + threadIdx.x;
  if (c >= F) return;
  int r0 = blockIdx.y * rowsPerBlk;
  int r1 = r0 + rowsPerBlk; if (r1 > Nrows) r1 = Nrows;
  float s = 0.f, s2 = 0.f;
  for (int r = r0; r < r1; ++r) {
    float v = x[(long long)r * F + c];
    s += v; s2 += v * v;
  }
  atomAddF(&sums[c], s);
  atomAddF(&sumsq[c], s2);
}

__global__ void k_bn_final(const float* __restrict__ sums, const float* __restrict__ sumsq,
                           const float* __restrict__ g, const float* __restrict__ beta,
                           float* __restrict__ scale, float* __restrict__ shift,
                           int Nrows, int F) {
  int c = blockIdx.x * blockDim.x + threadIdx.x;
  if (c >= F) return;
  float inv_n = 1.0f / (float)Nrows;
  float mean = sums[c] * inv_n;
  float var  = sumsq[c] * inv_n - mean * mean;  // biased variance
  float sc = g[c] * rsqrtf(var + BN_EPS);
  scale[c] = sc;
  shift[c] = beta[c] - mean * sc;
}

__global__ void k_bn_relu_bf16(const float* __restrict__ x, const float* __restrict__ scale,
                               const float* __restrict__ shift, bf16_t* __restrict__ y,
                               int F, long long total) {
  long long i  = (long long)blockIdx.x * blockDim.x + threadIdx.x;
  long long st = (long long)gridDim.x * blockDim.x;
  for (; i < total; i += st) {
    int c = (int)(i % F);
    float v = fmaxf(x[i] * scale[c] + shift[c], 0.0f);
    y[i] = (bf16_t)v;
  }
}

// ------------------------------- driver -----------------------------------

static inline int grid1d(long long total, int block, int cap) {
  long long g = (total + block - 1) / block;
  if (g > cap) g = cap;
  if (g < 1) g = 1;
  return (int)g;
}

extern "C" void kernel_launch(void* const* d_in, const int* in_sizes, int n_in,
                              void* d_out, int out_size, void* d_ws, size_t ws_size,
                              hipStream_t stream) {
  const int*   edge = (const int*)d_in[0];
  const float* z  = (const float*)d_in[1];
  const float* W1 = (const float*)d_in[2];  const float* b1 = (const float*)d_in[3];
  const float* g1 = (const float*)d_in[4];  const float* be1 = (const float*)d_in[5];
  const float* W2 = (const float*)d_in[6];  const float* b2 = (const float*)d_in[7];
  const float* g2 = (const float*)d_in[8];  const float* be2 = (const float*)d_in[9];
  const float* W3 = (const float*)d_in[10]; const float* b3 = (const float*)d_in[11];

  const int E  = in_sizes[0] / 2;
  const int IN = 128, HID = 256, OUT = 40, OUTP = 64;  // OUTP: padded layer-3 cols
  const int Nn = in_sizes[1] / IN;

  const int* srcv = edge;
  const int* dstv = edge + E;

  // ---- workspace carve-out ----
  char* ws = (char*)d_ws;
  auto carve = [&](size_t bytes) -> char* {
    char* p = ws;
    ws += (bytes + 255) & ~(size_t)255;
    return p;
  };
  float*  dinv  = (float*)carve((size_t)Nn * 4);
  bf16_t* zb    = (bf16_t*)carve((size_t)Nn * IN * 2);
  bf16_t* w1b   = (bf16_t*)carve((size_t)IN * HID * 2);
  bf16_t* w2b   = (bf16_t*)carve((size_t)HID * HID * 2);
  bf16_t* w3b   = (bf16_t*)carve((size_t)HID * OUTP * 2);   // padded
  float*  H     = (float*)carve((size_t)Nn * HID * 4);      // GEMM output
  float*  AGG   = (float*)carve((size_t)Nn * HID * 4);      // aggregation target
  bf16_t* XB    = (bf16_t*)carve((size_t)Nn * HID * 2);     // post-BN bf16 activations
  float*  sums  = (float*)carve((size_t)HID * 4);
  float*  sumsq = (float*)carve((size_t)HID * 4);
  float*  scale = (float*)carve((size_t)HID * 4);
  float*  shift = (float*)carve((size_t)HID * 4);

  const int BLK = 256;
  const int CAP = 1 << 20;

  // ---- normalization: deg (with self-loop) -> dinv ----
  k_fill_f32<<<grid1d(Nn, BLK, CAP), BLK, 0, stream>>>(dinv, 1.0f, Nn);
  k_deg<<<(E + BLK - 1) / BLK, BLK, 0, stream>>>(dstv, dinv, E);
  k_rsqrt_ip<<<(Nn + BLK - 1) / BLK, BLK, 0, stream>>>(dinv, Nn);

  // ---- bf16 conversions ----
  k_cvt_bf16<<<grid1d((long long)Nn * IN, BLK, CAP), BLK, 0, stream>>>(z, zb, (long long)Nn * IN);
  k_cvt_bf16<<<grid1d((long long)IN * HID, BLK, CAP), BLK, 0, stream>>>(W1, w1b, (long long)IN * HID);
  k_cvt_bf16<<<grid1d((long long)HID * HID, BLK, CAP), BLK, 0, stream>>>(W2, w2b, (long long)HID * HID);
  k_cvt_bf16_pad<<<grid1d((long long)HID * OUTP, BLK, CAP), BLK, 0, stream>>>(
      W3, w3b, OUT, OUTP, (long long)HID * OUTP);

  const int ROWS_PER_BLK = 512;
  const int rowBlks = (Nn + ROWS_PER_BLK - 1) / ROWS_PER_BLK;

  // ================= Layer 1: z(128) -> HID, BN+ReLU =================
  {
    long long tot = (long long)Nn * HID;
    k_bias_init<<<grid1d(tot, BLK, CAP), BLK, 0, stream>>>(b1, AGG, HID, tot);
    dim3 gg(HID / 64, Nn / 32);
    k_gemm_bf16_full<<<gg, BLK, 0, stream>>>(zb, w1b, H, IN, HID);
    long long aw = (long long)(E + Nn) * (HID / 4);
    k_aggregate<<<grid1d(aw, BLK, CAP), BLK, 0, stream>>>(H, srcv, dstv, dinv, AGG, HID, HID, E, Nn);
    k_fill_f32<<<1, BLK, 0, stream>>>(sums, 0.0f, HID);
    k_fill_f32<<<1, BLK, 0, stream>>>(sumsq, 0.0f, HID);
    dim3 sg((HID + BLK - 1) / BLK, rowBlks);
    k_bn_stats<<<sg, BLK, 0, stream>>>(AGG, sums, sumsq, Nn, HID, ROWS_PER_BLK);
    k_bn_final<<<1, BLK, 0, stream>>>(sums, sumsq, g1, be1, scale, shift, Nn, HID);
    k_bn_relu_bf16<<<grid1d(tot, BLK, CAP), BLK, 0, stream>>>(AGG, scale, shift, XB, HID, tot);
  }

  // ================= Layer 2: HID -> HID, BN+ReLU =================
  {
    long long tot = (long long)Nn * HID;
    k_bias_init<<<grid1d(tot, BLK, CAP), BLK, 0, stream>>>(b2, AGG, HID, tot);
    dim3 gg(HID / 64, Nn / 32);
    k_gemm_bf16_full<<<gg, BLK, 0, stream>>>(XB, w2b, H, HID, HID);
    long long aw = (long long)(E + Nn) * (HID / 4);
    k_aggregate<<<grid1d(aw, BLK, CAP), BLK, 0, stream>>>(H, srcv, dstv, dinv, AGG, HID, HID, E, Nn);
    k_fill_f32<<<1, BLK, 0, stream>>>(sums, 0.0f, HID);
    k_fill_f32<<<1, BLK, 0, stream>>>(sumsq, 0.0f, HID);
    dim3 sg((HID + BLK - 1) / BLK, rowBlks);
    k_bn_stats<<<sg, BLK, 0, stream>>>(AGG, sums, sumsq, Nn, HID, ROWS_PER_BLK);
    k_bn_final<<<1, BLK, 0, stream>>>(sums, sumsq, g2, be2, scale, shift, Nn, HID);
    k_bn_relu_bf16<<<grid1d(tot, BLK, CAP), BLK, 0, stream>>>(AGG, scale, shift, XB, HID, tot);
  }

  // ============ Layer 3: HID -> OUT(40) via padded 64, into d_out ============
  {
    float* outf = (float*)d_out;
    long long tot = (long long)Nn * OUT;
    k_bias_init<<<grid1d(tot, BLK, CAP), BLK, 0, stream>>>(b3, outf, OUT, tot);
    dim3 gg(OUTP / 64, Nn / 32);
    k_gemm_bf16_full<<<gg, BLK, 0, stream>>>(XB, w3b, H, HID, OUTP);  // H as N x 64
    long long aw = (long long)(E + Nn) * (OUT / 4);
    k_aggregate<<<grid1d(aw, BLK, CAP), BLK, 0, stream>>>(H, srcv, dstv, dinv, outf, OUT, OUTP, E, Nn);
  }
}
